// LSTM_24902220382916
// MI455X (gfx1250) — matrix operand order repaired and checked by reading, not compiled, not run on hardware
//
#include <hip/hip_runtime.h>

// ---------------------------------------------------------------------------
// LSTM for MI455X (gfx1250): bf16 WMMA (v_wmma_f32_16x16x32_bf16) with f32
// accumulate.
//   Phase 1: fused 4-gate input-projection GEMM (M=32768, N=1024, K=1024),
//            one wave per 16x16 tile, 4 WMMAs per k-step (A reused 4x).
//   Phase 2: 512 sequential step kernels. Each step fuses the 3 recurrent
//            GEMMs (M=64, N=1024, K=1024) with split-K=4 per tile + LDS
//            reduction, then the sigmoid/tanh cell update. Recurrent weights
//            (6 MB bf16) stay L2-resident across steps; step latency is the
//            bottleneck, so K is split to 8 WMMA-iterations per wave and
//            1024 waves run per step.
// ---------------------------------------------------------------------------

typedef __attribute__((ext_vector_type(16))) __bf16 v16bf;
typedef __attribute__((ext_vector_type(8)))  __bf16 v8bf;
typedef __attribute__((ext_vector_type(8)))  float  v8f;
typedef __attribute__((ext_vector_type(8)))  unsigned short v8us;

#define HDIM 1024
#define TLEN 512
#define BDIM 64
#define MROWS (BDIM * TLEN)   // 32768
#define KSPLIT 4
#define KCHUNK (HDIM / KSPLIT) // 256

// ---- helpers ---------------------------------------------------------------

__device__ __forceinline__ unsigned short f2bf(float f) {
    unsigned int u = __float_as_uint(f);
    u += 0x7FFFu + ((u >> 16) & 1u);   // round-to-nearest-even
    return (unsigned short)(u >> 16);
}
__device__ __forceinline__ float bf2f(unsigned short b) {
    return __uint_as_float(((unsigned int)b) << 16);
}
__device__ __forceinline__ v8bf ld8bf(const unsigned short* p) {
    return *reinterpret_cast<const v8bf*>(p);   // 16B aligned -> global_load_b128
}
__device__ __forceinline__ v16bf mk16(v8bf lo, v8bf hi) {
    v16bf r;
#pragma unroll
    for (int i = 0; i < 8; ++i) { r[i] = lo[i]; r[i + 8] = hi[i]; }
    return r;
}
__device__ __forceinline__ v8f wmma_bf16(v16bf a, v16bf b, v8f c) {
    return __builtin_amdgcn_wmma_f32_16x16x32_bf16(
        /*neg_a=*/false, a, /*neg_b=*/false, b,
        /*c_mod=*/(short)0, c, /*reuse_a=*/false, /*reuse_b=*/false);
}
__device__ __forceinline__ float sigmoidf_fast(float x) {
    return 1.0f / (1.0f + __expf(-x));
}
__device__ __forceinline__ float tanhf_fast(float x) {
    float e = __expf(2.0f * x);
    return (e - 1.0f) / (e + 1.0f);
}

// ---- conversion / init kernels --------------------------------------------

__global__ __launch_bounds__(256) void cvt_f32_bf16_x8(
    const float* __restrict__ in, unsigned short* __restrict__ out, int n8) {
    int i = blockIdx.x * blockDim.x + threadIdx.x;
    if (i >= n8) return;
    const float4* p = reinterpret_cast<const float4*>(in) + (size_t)2 * i;
    float4 a = p[0], b = p[1];
    v8us o;
    o[0] = f2bf(a.x); o[1] = f2bf(a.y); o[2] = f2bf(a.z); o[3] = f2bf(a.w);
    o[4] = f2bf(b.x); o[5] = f2bf(b.y); o[6] = f2bf(b.z); o[7] = f2bf(b.w);
    *(reinterpret_cast<v8us*>(out) + i) = o;   // global_store_b128
}

__global__ __launch_bounds__(256) void init_cell_state(
    const float* __restrict__ c0, float* __restrict__ ctf,
    unsigned short* __restrict__ ctb) {
    int i = blockIdx.x * blockDim.x + threadIdx.x;   // B*H = 65536
    if (i >= BDIM * HDIM) return;
    float v = c0[i & (HDIM - 1)];
    ctf[i] = v;
    ctb[i] = f2bf(v);
}

// ---- phase 1: fused 4-gate input projection GEMM ---------------------------
// One wave per 16x16 output tile; 4 WMMAs per k-step (x-tile reused 4x).
// Tiles: 2048 (M) x 64 (N) = 131072 waves = 16384 blocks of 8 waves.

__global__ __launch_bounds__(256) void lstm_gate_gemm(
    const unsigned short* __restrict__ xb,
    const unsigned short* __restrict__ Wf, const unsigned short* __restrict__ Wi,
    const unsigned short* __restrict__ Wo, const unsigned short* __restrict__ Wc,
    const float* __restrict__ bxc,
    unsigned short* __restrict__ gf, unsigned short* __restrict__ gi,
    unsigned short* __restrict__ go, unsigned short* __restrict__ gc) {
    int wave = (blockIdx.x * blockDim.x + threadIdx.x) >> 5;
    int lane = threadIdx.x & 31;
    int nt = wave & 63;           // N tile 0..63
    int mt = wave >> 6;           // M tile 0..2047
    int l  = lane & 15;
    int hi = lane >> 4;

    const unsigned short* arow = xb + (size_t)(mt * 16 + l) * HDIM;
    size_t woff = (size_t)(nt * 16 + l) * HDIM + hi * 16;
    const unsigned short* pf = Wf + woff;
    const unsigned short* pi = Wi + woff;
    const unsigned short* po = Wo + woff;
    const unsigned short* pc = Wc + woff;
    int alo = hi * 16;

    v8f accf = {}, acci = {}, acco = {}, accc = {};
    for (int k = 0; k < HDIM; k += 32) {
        v16bf a  = mk16(ld8bf(arow + k + alo), ld8bf(arow + k + alo + 8));
        v16bf bf = mk16(ld8bf(pf + k), ld8bf(pf + k + 8));
        v16bf bi = mk16(ld8bf(pi + k), ld8bf(pi + k + 8));
        v16bf bo = mk16(ld8bf(po + k), ld8bf(po + k + 8));
        v16bf bc = mk16(ld8bf(pc + k), ld8bf(pc + k + 8));
        accf = wmma_bf16(a, bf, accf);
        acci = wmma_bf16(a, bi, acci);
        acco = wmma_bf16(a, bo, acco);
        accc = wmma_bf16(a, bc, accc);
    }

    int n = nt * 16 + l;
    float bias_c = bxc[n];
#pragma unroll
    for (int j = 0; j < 8; ++j) {
        size_t m = (size_t)mt * 16 + j + hi * 8;
        size_t idx = m * HDIM + n;
        gf[idx] = f2bf(accf[j]);
        gi[idx] = f2bf(acci[j]);
        go[idx] = f2bf(acco[j]);
        gc[idx] = f2bf(accc[j] + bias_c);
    }
}

// ---- phase 2: one recurrent step (split-K=4 + LDS reduction) ---------------
// 256 tiles (4 M x 64 N) x 4 K-chunks = 1024 waves = 128 blocks of 8 waves.
// Block = 2 tiles x 4 chunks; chunk-0 wave reduces partials and runs the
// gate epilogue for its tile.

__global__ __launch_bounds__(256) void lstm_step(
    const unsigned short* __restrict__ ctb,   // prev cell, bf16 [B,H]
    const float* __restrict__ ctf,            // prev cell, f32  [B,H]
    const unsigned short* __restrict__ Wcf, const unsigned short* __restrict__ Wci,
    const unsigned short* __restrict__ Wco,
    const float* __restrict__ bcf, const float* __restrict__ bci,
    const float* __restrict__ bco,
    const unsigned short* __restrict__ gf, const unsigned short* __restrict__ gi,
    const unsigned short* __restrict__ go, const unsigned short* __restrict__ gc,
    float* __restrict__ ct_new_f, unsigned short* __restrict__ ct_new_b,
    float* __restrict__ hidden,               // d_out [B,T,H]
    float* __restrict__ hlast,                // d_out tail [B,H]
    int t) {
    __shared__ float red[8][3][32][8];        // [wave][gate][lane][j] = 24 KB

    int w    = threadIdx.x >> 5;              // wave in block: 0..7
    int lane = threadIdx.x & 31;
    int tileLocal = w >> 2;                   // 0..1
    int kc        = w & (KSPLIT - 1);         // K-chunk 0..3
    int tile = blockIdx.x * 2 + tileLocal;    // 0..255
    int mt = tile & 3;                        // M tile 0..3  (B=64)
    int nt = tile >> 2;                       // N tile 0..63 (H=1024)
    int l  = lane & 15;
    int hi = lane >> 4;

    const unsigned short* arow = ctb + (size_t)(mt * 16 + l) * HDIM;
    size_t woff = (size_t)(nt * 16 + l) * HDIM + hi * 16;
    const unsigned short* pf = Wcf + woff;
    const unsigned short* pi = Wci + woff;
    const unsigned short* po = Wco + woff;
    int alo = hi * 16;

    v8f accf = {}, acci = {}, acco = {};
    int k0 = kc * KCHUNK;
    for (int k = k0; k < k0 + KCHUNK; k += 32) {
        v16bf a  = mk16(ld8bf(arow + k + alo), ld8bf(arow + k + alo + 8));
        v16bf bf = mk16(ld8bf(pf + k), ld8bf(pf + k + 8));
        v16bf bi = mk16(ld8bf(pi + k), ld8bf(pi + k + 8));
        v16bf bo = mk16(ld8bf(po + k), ld8bf(po + k + 8));
        accf = wmma_bf16(a, bf, accf);
        acci = wmma_bf16(a, bi, acci);
        acco = wmma_bf16(a, bo, acco);
    }

    // stash partial accumulators (32B per lane per gate -> ds_store_b128 x2)
#pragma unroll
    for (int j = 0; j < 8; ++j) {
        red[w][0][lane][j] = accf[j];
        red[w][1][lane][j] = acci[j];
        red[w][2][lane][j] = acco[j];
    }
    __syncthreads();
    if (kc != 0) return;

    int wb = tileLocal * KSPLIT;
#pragma unroll
    for (int j = 0; j < 8; ++j) {
        accf[j] = red[wb][0][lane][j] + red[wb + 1][0][lane][j]
                + red[wb + 2][0][lane][j] + red[wb + 3][0][lane][j];
        acci[j] = red[wb][1][lane][j] + red[wb + 1][1][lane][j]
                + red[wb + 2][1][lane][j] + red[wb + 3][1][lane][j];
        acco[j] = red[wb][2][lane][j] + red[wb + 1][2][lane][j]
                + red[wb + 2][2][lane][j] + red[wb + 3][2][lane][j];
    }

    int n = nt * 16 + l;
    float bf_n = bcf[n], bi_n = bci[n], bo_n = bco[n];
#pragma unroll
    for (int j = 0; j < 8; ++j) {
        int b = mt * 16 + j + hi * 8;                       // batch row 0..63
        size_t gidx = ((size_t)b * TLEN + t) * HDIM + n;    // gate/output index
        size_t sidx = (size_t)b * HDIM + n;                 // state index
        float f = sigmoidf_fast(accf[j] + bf2f(gf[gidx]) + bf_n);
        float i = sigmoidf_fast(acci[j] + bf2f(gi[gidx]) + bi_n);
        float o = sigmoidf_fast(acco[j] + bf2f(go[gidx]) + bo_n);
        float cn = f * ctf[sidx] + i * tanhf_fast(bf2f(gc[gidx]));
        float h = o * tanhf_fast(cn);
        ct_new_f[sidx] = cn;
        ct_new_b[sidx] = f2bf(cn);
        hidden[gidx] = h;
        if (t == TLEN - 1) hlast[sidx] = h;
    }
}

// ---- host ------------------------------------------------------------------

extern "C" void kernel_launch(void* const* d_in, const int* in_sizes, int n_in,
                              void* d_out, int out_size, void* d_ws, size_t ws_size,
                              hipStream_t stream) {
    const float* x   = (const float*)d_in[0];
    const float* Wxf = (const float*)d_in[1];
    const float* Wcf = (const float*)d_in[2];
    const float* bcf = (const float*)d_in[3];
    const float* Wxi = (const float*)d_in[4];
    const float* Wci = (const float*)d_in[5];
    const float* bci = (const float*)d_in[6];
    const float* Wxo = (const float*)d_in[7];
    const float* Wco = (const float*)d_in[8];
    const float* bco = (const float*)d_in[9];
    const float* Wxc = (const float*)d_in[10];
    const float* bxc = (const float*)d_in[11];
    const float* c0  = (const float*)d_in[12];

    float* hidden = (float*)d_out;                          // [B,T,H]
    float* hlast  = hidden + (size_t)BDIM * TLEN * HDIM;    // [B,H]

    // workspace layout (~336 MB)
    char* base = (char*)d_ws;
    size_t off = 0;
    auto take = [&](size_t bytes) {
        char* p = base + off;
        off = (off + bytes + 255) & ~(size_t)255;
        return p;
    };
    unsigned short* xb    = (unsigned short*)take((size_t)MROWS * HDIM * 2);
    unsigned short* Wxf_b = (unsigned short*)take((size_t)HDIM * HDIM * 2);
    unsigned short* Wxi_b = (unsigned short*)take((size_t)HDIM * HDIM * 2);
    unsigned short* Wxo_b = (unsigned short*)take((size_t)HDIM * HDIM * 2);
    unsigned short* Wxc_b = (unsigned short*)take((size_t)HDIM * HDIM * 2);
    unsigned short* Wcf_b = (unsigned short*)take((size_t)HDIM * HDIM * 2);
    unsigned short* Wci_b = (unsigned short*)take((size_t)HDIM * HDIM * 2);
    unsigned short* Wco_b = (unsigned short*)take((size_t)HDIM * HDIM * 2);
    unsigned short* gf = (unsigned short*)take((size_t)MROWS * HDIM * 2);
    unsigned short* gi = (unsigned short*)take((size_t)MROWS * HDIM * 2);
    unsigned short* go = (unsigned short*)take((size_t)MROWS * HDIM * 2);
    unsigned short* gc = (unsigned short*)take((size_t)MROWS * HDIM * 2);
    float* ctf0          = (float*)take((size_t)BDIM * HDIM * 4);
    float* ctf1          = (float*)take((size_t)BDIM * HDIM * 4);
    unsigned short* ctb0 = (unsigned short*)take((size_t)BDIM * HDIM * 2);
    unsigned short* ctb1 = (unsigned short*)take((size_t)BDIM * HDIM * 2);

    // 1) convert x and weights to bf16
    {
        int n8 = (MROWS * HDIM) / 8;   // 4,194,304
        cvt_f32_bf16_x8<<<n8 / 256, 256, 0, stream>>>(x, xb, n8);
        int w8 = (HDIM * HDIM) / 8;    // 131,072
        cvt_f32_bf16_x8<<<w8 / 256, 256, 0, stream>>>(Wxf, Wxf_b, w8);
        cvt_f32_bf16_x8<<<w8 / 256, 256, 0, stream>>>(Wxi, Wxi_b, w8);
        cvt_f32_bf16_x8<<<w8 / 256, 256, 0, stream>>>(Wxo, Wxo_b, w8);
        cvt_f32_bf16_x8<<<w8 / 256, 256, 0, stream>>>(Wxc, Wxc_b, w8);
        cvt_f32_bf16_x8<<<w8 / 256, 256, 0, stream>>>(Wcf, Wcf_b, w8);
        cvt_f32_bf16_x8<<<w8 / 256, 256, 0, stream>>>(Wci, Wci_b, w8);
        cvt_f32_bf16_x8<<<w8 / 256, 256, 0, stream>>>(Wco, Wco_b, w8);
    }

    // 2) init cell state from c0 (broadcast over batch)
    init_cell_state<<<(BDIM * HDIM) / 256, 256, 0, stream>>>(c0, ctf0, ctb0);

    // 3) phase 1: fused input-projection GEMM (131072 tiles, 8 waves/block)
    lstm_gate_gemm<<<16384, 256, 0, stream>>>(xb, Wxf_b, Wxi_b, Wxo_b, Wxc_b,
                                              bxc, gf, gi, go, gc);

    // 4) phase 2: 512 dependent steps, ping-pong cell-state buffers
    for (int t = 0; t < TLEN; ++t) {
        const bool even = (t & 1) == 0;
        lstm_step<<<128, 256, 0, stream>>>(
            even ? ctb0 : ctb1, even ? ctf0 : ctf1,
            Wcf_b, Wci_b, Wco_b, bcf, bci, bco,
            gf, gi, go, gc,
            even ? ctf1 : ctf0, even ? ctb1 : ctb0,
            hidden, hlast, t);
    }
}
